// SelfAttention_79113297592362
// MI455X (gfx1250) — compile-verified, matrix operand order checked by reading, not run on hardware
//
#include <hip/hip_runtime.h>
#include <hip/hip_bf16.h>

// ---------------------------------------------------------------------------
// Fused self-attention (no softmax) for MI455X / gfx1250, wave32 + WMMA.
//   q = x@Wq^T + bq ; k = x@Wk^T + bk ; v = x@Wv^T + bv      (f16, f32 accum)
//   out = q @ (k^T @ v) / sqrt(H)        <-- associativity: no NxN scores
// All GEMM operands are K-contiguous f16 rows -> uniform b128 tile loader,
// double-buffered LDS, async global->LDS copies (ASYNCcnt), 64x64 wave tiles.
// ---------------------------------------------------------------------------

typedef __attribute__((ext_vector_type(16))) _Float16 v16h;
typedef __attribute__((ext_vector_type(8)))  _Float16 h8;
typedef __attribute__((ext_vector_type(4)))  _Float16 h4;
typedef __attribute__((ext_vector_type(8)))  float    v8f;
typedef __attribute__((ext_vector_type(4)))  float    f32x4;
typedef __attribute__((ext_vector_type(4)))  int      v4i;

constexpr int BM = 256;       // block tile M (4 waves x 64)
constexpr int BN = 128;       // block tile N (2 waves x 64)
constexpr int BK = 32;        // K step (matches 16x16x32 WMMA)
constexpr int LDST = 40;      // LDS row stride in halfwords (80B = 5*16B)

constexpr int Nrows = 8192;
constexpr int Hdim  = 1024;
constexpr int KSPLIT = 8;     // deterministic split-K for k^T @ v

// ---- gfx1250 async copy-to-LDS detection ----------------------------------
#define AS3 __attribute__((address_space(3)))
#define AS1 __attribute__((address_space(1)))
#if defined(__has_builtin)
#  if __has_builtin(__builtin_amdgcn_global_load_async_to_lds_b128) && \
      __has_builtin(__builtin_amdgcn_s_wait_asynccnt)
#    define USE_ASYNC_COPY 1
#  endif
#endif
#ifndef USE_ASYNC_COPY
#  define USE_ASYNC_COPY 0
#endif

#if USE_ASYNC_COPY
__device__ __forceinline__ AS3 v4i* lds_cast(void* p) {
  return (AS3 v4i*)(unsigned int)(unsigned long long)p;
}
__device__ __forceinline__ AS1 v4i* gbl_cast(const void* p) {
  return (AS1 v4i*)(unsigned long long)p;
}
#endif

// ---- issue one ROWSx32 f16 tile: global -> LDS (b128 per transfer) --------
template <int ROWS>
__device__ __forceinline__ void issue_tile(_Float16* __restrict__ lds,
                                           const _Float16* __restrict__ g,
                                           int ld, int row0, int kpos, int tid) {
#pragma unroll
  for (int p = 0; p < ROWS / 64; ++p) {      // ROWS*4 x 16B transfers / 256 thr
    int idx = p * 256 + tid;
    int r   = idx >> 2;
    int c8  = (idx & 3) * 8;
    const _Float16* gp = g + (size_t)(row0 + r) * ld + (kpos + c8);
    _Float16*       lp = lds + r * LDST + c8;        // 16B-aligned
#if USE_ASYNC_COPY
    __builtin_amdgcn_global_load_async_to_lds_b128(gbl_cast(gp), lds_cast(lp),
                                                   0, 0);
#else
    *reinterpret_cast<h8*>(lp) = *reinterpret_cast<const h8*>(gp);
#endif
  }
}

// ---- per-lane WMMA fragment from a [ROWS][LDST] LDS tile -------------------
// 16-bit operand layout (ISA 7.12.2): lanes 0-15 hold K {0..7,16..23},
// lanes 16-31 hold K {8..15,24..31}; row/col = lane & 15.
__device__ __forceinline__ v16h load_frag(const _Float16* __restrict__ lds,
                                          int rowbase, int lane) {
  int r  = rowbase + (lane & 15);
  int kb = (lane >> 4) << 3;                 // 0 or 8
  const _Float16* p = lds + r * LDST + kb;   // 16-byte aligned chunks
  v16h f;
#pragma unroll
  for (int i = 0; i < 8; ++i) { f[i] = p[i]; f[i + 8] = p[i + 16]; }
  return f;
}

// ---- all-f16 WMMA GEMM:  C[m,n] = sum_k A[m,k]*B[n,k] (+ bias[n]) ----------
// STORET: store C transposed (C_T[n*ldc + m], f16) -> one b128 store per tile.
template <bool STORET, bool BIAS, typename TOut>
__global__ __launch_bounds__(256) void gemm_f16_wmma(
    const _Float16* __restrict__ A, const _Float16* __restrict__ B,
    const float* __restrict__ bias, TOut* __restrict__ C,
    int K, int lda, int ldb, int ldc) {
  __shared__ __align__(16) _Float16 lsA[2][BM * LDST];
  __shared__ __align__(16) _Float16 lsB[2][BN * LDST];

  const int tid  = threadIdx.x;
  const int lane = tid & 31;
  const int wid  = tid >> 5;       // 8 waves
  const int wm   = wid & 3;        // 4 waves along M (64 rows each)
  const int wn   = wid >> 2;       // 2 waves along N (64 cols each)
  const int row0 = blockIdx.y * BM;
  const int col0 = blockIdx.x * BN;

  v8f acc[4][4] = {};              // 64x64 per wave = 16 WMMA accumulators

  const int nk = K / BK;
  issue_tile<BM>(lsA[0], A, lda, row0, 0, tid);
  issue_tile<BN>(lsB[0], B, ldb, col0, 0, tid);

  for (int kt = 0; kt < nk; ++kt) {
    const int cur = kt & 1;
    if (kt + 1 < nk) {
      issue_tile<BM>(lsA[cur ^ 1], A, lda, row0, (kt + 1) * BK, tid);
      issue_tile<BN>(lsB[cur ^ 1], B, ldb, col0, (kt + 1) * BK, tid);
#if USE_ASYNC_COPY
      __builtin_amdgcn_s_wait_asynccnt(6);   // allow next tile-pair in flight
#endif
    } else {
#if USE_ASYNC_COPY
      __builtin_amdgcn_s_wait_asynccnt(0);
#endif
    }
    if (kt + 2 < nk) {                       // L2 prefetch two tiles ahead
      __builtin_prefetch(A + (size_t)(row0 + tid) * lda + (kt + 2) * BK, 0, 0);
      __builtin_prefetch(B + (size_t)(col0 + (tid & 127)) * ldb +
                             (kt + 2) * BK, 0, 0);
    }
    __syncthreads();                         // cur buffers fully landed

    v16h af[4], bf[4];
#pragma unroll
    for (int i = 0; i < 4; ++i)
      af[i] = load_frag(lsA[cur], wm * 64 + i * 16, lane);
#pragma unroll
    for (int j = 0; j < 4; ++j)
      bf[j] = load_frag(lsB[cur], wn * 64 + j * 16, lane);

#pragma unroll
    for (int i = 0; i < 4; ++i)
#pragma unroll
      for (int j = 0; j < 4; ++j)
        acc[i][j] = __builtin_amdgcn_wmma_f32_16x16x32_f16(
            false, af[i], false, bf[j], (short)0, acc[i][j], false, false);

    __syncthreads();                         // done reading cur before refill
  }

  // C/D layout (ISA 7.12.2): VGPR g -> M = g + 8*(lane>=16), N = lane&15.
#pragma unroll
  for (int i = 0; i < 4; ++i)
#pragma unroll
    for (int j = 0; j < 4; ++j) {
      int gm = row0 + wm * 64 + i * 16 + ((lane >> 4) << 3);
      int gn = col0 + wn * 64 + j * 16 + (lane & 15);
      float bv = 0.0f;
      if constexpr (BIAS) bv = bias[gn];
      if constexpr (STORET) {
        h8 hv;                               // 8 consecutive m at fixed n
#pragma unroll
        for (int gg = 0; gg < 8; ++gg)
          hv[gg] = (_Float16)(acc[i][j][gg] + bv);
        *reinterpret_cast<h8*>((_Float16*)C + (size_t)gn * ldc + gm) = hv;
      } else {
#pragma unroll
        for (int gg = 0; gg < 8; ++gg) {
          float val = acc[i][j][gg] + bv;
          C[(size_t)(gm + gg) * ldc + gn] = (TOut)val;
        }
      }
    }
}

// ---- fp32 -> f16 convert (vectorized, exact sizes) -------------------------
__global__ __launch_bounds__(256) void cvt_f32_f16(const float* __restrict__ src,
                                                   _Float16* __restrict__ dst) {
  size_t i = (size_t)blockIdx.x * 256 + threadIdx.x;   // index of float4
  f32x4 v = reinterpret_cast<const f32x4*>(src)[i];
  reinterpret_cast<h4*>(dst)[i] = __builtin_convertvector(v, h4);
}

// ---- reduce split-K partials of KV, scale by 1/sqrt(H), store KV^T (f16) ---
__global__ __launch_bounds__(256) void reduce_kv(const float* __restrict__ P,
                                                 _Float16* __restrict__ KVt,
                                                 float scale) {
  int idx = blockIdx.x * 256 + threadIdx.x;     // 0 .. H*H-1
  int h1 = idx >> 10;
  int h2 = idx & (Hdim - 1);
  float s = 0.0f;
#pragma unroll
  for (int z = 0; z < KSPLIT; ++z)
    s += P[(size_t)z * Hdim * Hdim + idx];
  KVt[(size_t)h2 * Hdim + h1] = (_Float16)(s * scale);   // transposed store
}

// ---------------------------------------------------------------------------
extern "C" void kernel_launch(void* const* d_in, const int* in_sizes, int n_in,
                              void* d_out, int out_size, void* d_ws, size_t ws_size,
                              hipStream_t stream) {
  (void)in_sizes; (void)n_in; (void)out_size; (void)ws_size;
  const float* x  = (const float*)d_in[0];
  const float* Wq = (const float*)d_in[1];
  const float* bq = (const float*)d_in[2];
  const float* Wk = (const float*)d_in[3];
  const float* bk = (const float*)d_in[4];
  const float* Wv = (const float*)d_in[5];
  const float* bv = (const float*)d_in[6];
  float* out = (float*)d_out;

  char* ws = (char*)d_ws;
  const size_t MB = 1ull << 20;
  _Float16* xf  = (_Float16*)(ws);               // 16 MB
  _Float16* Wqf = (_Float16*)(ws + 16 * MB);     //  2 MB
  _Float16* Wkf = (_Float16*)(ws + 18 * MB);     //  2 MB
  _Float16* Wvf = (_Float16*)(ws + 20 * MB);     //  2 MB
  _Float16* qf  = (_Float16*)(ws + 22 * MB);     // 16 MB   q   [N,H]
  _Float16* kT  = (_Float16*)(ws + 38 * MB);     // 16 MB   k^T [H,N]
  _Float16* vT  = (_Float16*)(ws + 54 * MB);     // 16 MB   v^T [H,N]
  float*    P   = (float*)   (ws + 70 * MB);     // 32 MB   8 partials [H,H]
  _Float16* KVt = (_Float16*)(ws + 102 * MB);    //  2 MB   KV^T [H,H]

  dim3 blk(256);

  // fp32 -> f16 converts
  cvt_f32_f16<<<(Nrows * Hdim) / 1024, blk, 0, stream>>>(x, xf);
  cvt_f32_f16<<<(Hdim * Hdim) / 1024, blk, 0, stream>>>(Wq, Wqf);
  cvt_f32_f16<<<(Hdim * Hdim) / 1024, blk, 0, stream>>>(Wk, Wkf);
  cvt_f32_f16<<<(Hdim * Hdim) / 1024, blk, 0, stream>>>(Wv, Wvf);

  dim3 gProj(Hdim / BN, Nrows / BM);   // 8 x 32
  dim3 gKV(Hdim / BN, Hdim / BM);      // 8 x 4 (x 8 split-K launches)

  // projections: q normal layout; k,v stored transposed ([H, N], ldc = N)
  gemm_f16_wmma<false, true, _Float16>
      <<<gProj, blk, 0, stream>>>(xf, Wqf, bq, qf, Hdim, Hdim, Hdim, Hdim);
  gemm_f16_wmma<true, true, _Float16>
      <<<gProj, blk, 0, stream>>>(xf, Wkf, bk, kT, Hdim, Hdim, Hdim, Nrows);
  gemm_f16_wmma<true, true, _Float16>
      <<<gProj, blk, 0, stream>>>(xf, Wvf, bv, vT, Hdim, Hdim, Hdim, Nrows);

  // KV partials: KV[h1,h2] = sum_n kT[h1,n] * vT[h2,n]  (both row-major now)
  for (int z = 0; z < KSPLIT; ++z)
    gemm_f16_wmma<false, false, float>
        <<<gKV, blk, 0, stream>>>(kT + (size_t)z * (Nrows / KSPLIT),
                                  vT + (size_t)z * (Nrows / KSPLIT), nullptr,
                                  P + (size_t)z * Hdim * Hdim,
                                  Nrows / KSPLIT, Nrows, Nrows, Hdim);

  // sum partials, scale by 1/sqrt(1024), emit KV^T in f16
  reduce_kv<<<(Hdim * Hdim) / 256, blk, 0, stream>>>(P, KVt, 0.03125f);

  // out = q @ KV  (KVt rows are KV columns -> contiguous B fragments)
  gemm_f16_wmma<false, false, float>
      <<<gProj, blk, 0, stream>>>(qf, KVt, nullptr, out, Hdim, Hdim, Hdim, Hdim);
}
